// TimeMixing_2319282340089
// MI455X (gfx1250) — compile-verified, matrix operand order checked by reading, not run on hardware
//
#include <hip/hip_runtime.h>
#include <cstdint>

// ---------------------------------------------------------------- types
typedef __attribute__((ext_vector_type(16))) __bf16 bf16x16;
typedef __attribute__((ext_vector_type(8)))  __bf16 bf16x8;
typedef __attribute__((ext_vector_type(8)))  float  f32x8;

static constexpr int  Bdim = 4, Tdim = 1024, Ddim = 1024, Hdim = 16, HDs = 64, LR = 64;
static constexpr int  Ntok = Bdim * Tdim;                    // 4096 token rows
static constexpr long ND   = (long)Ntok * Ddim;              // 4 Mi elements
static constexpr long NL   = (long)Ntok * LR;
static constexpr long DDsz = (long)Ddim * Ddim;
static constexpr long LDsz = (long)LR * Ddim;

__device__ __forceinline__ __bf16 f2bf(float f) {
  union { float f; uint32_t u; } x; x.f = f;
  uint32_t r = x.u + 0x7fffu + ((x.u >> 16) & 1u);   // RNE
  uint16_t hs = (uint16_t)(r >> 16);
  __bf16 b; __builtin_memcpy(&b, &hs, 2);
  return b;
}
__device__ __forceinline__ float bf2f(__bf16 b) {
  uint16_t hs; __builtin_memcpy(&hs, &b, 2);
  union { uint32_t u; float f; } x; x.u = ((uint32_t)hs) << 16;
  return x.f;
}

// ---------------------------------------------------------------- prep: token shift + lerp
__global__ void tm_prep(const float* __restrict__ x, const float* __restrict__ xw,
                        __bf16* __restrict__ xb, __bf16* __restrict__ dxb,
                        __bf16* __restrict__ lerpb) {
  long idx = (long)blockIdx.x * blockDim.x + threadIdx.x;
  if (idx >= ND) return;
  int  c = (int)(idx % Ddim);
  long n = idx / Ddim;
  int  t = (int)(n % Tdim);
  float xv = x[idx];
  float xx = (t > 0) ? x[idx - Ddim] : 0.f;
  float dx = xx - xv;
  xb[idx]    = f2bf(xv);
  dxb[idx]   = f2bf(dx);
  lerpb[idx] = f2bf(xv + dx * xw[c]);
}

// ---------------------------------------------------------------- f32 -> bf16
__global__ void tm_f2bf(const float* __restrict__ src, __bf16* __restrict__ dst, long n) {
  long i = (long)blockIdx.x * blockDim.x + threadIdx.x;
  if (i < n) dst[i] = f2bf(src[i]);
}

// ---------------------------------------------------------------- generic WMMA GEMM: C = A @ W^T
// A: [M,K] bf16 row-major, W: [N,K] bf16 row-major.
// Wave tile = 32x64: 2 A frags + 4 B frags -> 8 WMMAs per 32-K step (1.5 b128 loads / WMMA).
enum { MODE_TANH_BF16 = 0, MODE_MIX_BF16 = 1, MODE_W_F32 = 2, MODE_F32 = 3 };

__global__ void __launch_bounds__(256)
tm_wmma_gemm(const __bf16* __restrict__ A, int lda,
             const __bf16* __restrict__ W, int ldw,
             int M, int N, int K, int mode,
             float* __restrict__ outF, __bf16* __restrict__ outB,
             const __bf16* __restrict__ xb, const __bf16* __restrict__ dxb,
             const float* __restrict__ ll) {
  const int lane   = threadIdx.x & 31;
  const int wave   = threadIdx.x >> 5;
  const int mtiles = M >> 5;                    // 32-row wave tiles
  const int total  = mtiles * (N >> 6);
  const int row16  = lane & 15;
  const int koff   = (lane >> 4) << 3;          // 0 or 8 (ISA bf16 frag K-chunk base)

  for (int wt = blockIdx.x * 8 + wave; wt < total; wt += gridDim.x * 8) {
    const int m0 = (wt % mtiles) << 5;
    const int n0 = (wt / mtiles) << 6;

    f32x8 acc[2][4];
    #pragma unroll
    for (int mt = 0; mt < 2; ++mt)
      #pragma unroll
      for (int nt = 0; nt < 4; ++nt) acc[mt][nt] = (f32x8){};

    const __bf16* a0r = A + (size_t)(m0 +  0 + row16) * lda + koff;
    const __bf16* a1r = A + (size_t)(m0 + 16 + row16) * lda + koff;
    const __bf16* w0r = W + (size_t)(n0 +  0 + row16) * ldw + koff;
    const __bf16* w1r = W + (size_t)(n0 + 16 + row16) * ldw + koff;
    const __bf16* w2r = W + (size_t)(n0 + 32 + row16) * ldw + koff;
    const __bf16* w3r = W + (size_t)(n0 + 48 + row16) * ldw + koff;

    for (int k = 0; k < K; k += 32) {
      union { bf16x16 v; bf16x8 h[2]; } a0, a1, b0, b1, b2, b3;
      a0.h[0] = *(const bf16x8*)(a0r + k); a0.h[1] = *(const bf16x8*)(a0r + k + 16);
      a1.h[0] = *(const bf16x8*)(a1r + k); a1.h[1] = *(const bf16x8*)(a1r + k + 16);
      b0.h[0] = *(const bf16x8*)(w0r + k); b0.h[1] = *(const bf16x8*)(w0r + k + 16);
      b1.h[0] = *(const bf16x8*)(w1r + k); b1.h[1] = *(const bf16x8*)(w1r + k + 16);
      b2.h[0] = *(const bf16x8*)(w2r + k); b2.h[1] = *(const bf16x8*)(w2r + k + 16);
      b3.h[0] = *(const bf16x8*)(w3r + k); b3.h[1] = *(const bf16x8*)(w3r + k + 16);

      if (k + 32 < K) {                           // gfx1250 global_prefetch_b8 on next chunk
        __builtin_prefetch(a0r + k + 32, 0, 0);
        __builtin_prefetch(a1r + k + 32, 0, 0);
        __builtin_prefetch(w0r + k + 32, 0, 0);
        __builtin_prefetch(w2r + k + 32, 0, 0);
      }

      acc[0][0] = __builtin_amdgcn_wmma_f32_16x16x32_bf16(false, a0.v, false, b0.v, (short)0, acc[0][0], false, false);
      acc[0][1] = __builtin_amdgcn_wmma_f32_16x16x32_bf16(false, a0.v, false, b1.v, (short)0, acc[0][1], false, false);
      acc[0][2] = __builtin_amdgcn_wmma_f32_16x16x32_bf16(false, a0.v, false, b2.v, (short)0, acc[0][2], false, false);
      acc[0][3] = __builtin_amdgcn_wmma_f32_16x16x32_bf16(false, a0.v, false, b3.v, (short)0, acc[0][3], false, false);
      acc[1][0] = __builtin_amdgcn_wmma_f32_16x16x32_bf16(false, a1.v, false, b0.v, (short)0, acc[1][0], false, false);
      acc[1][1] = __builtin_amdgcn_wmma_f32_16x16x32_bf16(false, a1.v, false, b1.v, (short)0, acc[1][1], false, false);
      acc[1][2] = __builtin_amdgcn_wmma_f32_16x16x32_bf16(false, a1.v, false, b2.v, (short)0, acc[1][2], false, false);
      acc[1][3] = __builtin_amdgcn_wmma_f32_16x16x32_bf16(false, a1.v, false, b3.v, (short)0, acc[1][3], false, false);
    }

    #pragma unroll
    for (int mt = 0; mt < 2; ++mt) {
      const int mb = m0 + mt * 16 + ((lane >> 4) << 3);   // lanes 16-31 hold rows +8..+15
      #pragma unroll
      for (int nt = 0; nt < 4; ++nt) {
        const int col = n0 + nt * 16 + row16;
        #pragma unroll
        for (int r = 0; r < 8; ++r) {
          const size_t idx = (size_t)(mb + r) * N + col;
          const float v = acc[mt][nt][r];
          if (mode == MODE_TANH_BF16) {
            outB[idx] = f2bf(tanhf(v));
          } else if (mode == MODE_MIX_BF16) {
            outB[idx] = f2bf(bf2f(xb[idx]) + bf2f(dxb[idx]) * (ll[col] + v));
          } else if (mode == MODE_W_F32) {
            outF[idx] = expf(-expf(ll[col] + v));
          } else {
            outF[idx] = v;
          }
        }
      }
    }
  }
}

// ---------------------------------------------------------------- sequential scan over T
// grid = B*H blocks; thread (iz,j): owns S[iz*16 .. iz*16+15][j] in registers.
__global__ void __launch_bounds__(256)
tm_scan(const float* __restrict__ r, const float* __restrict__ w,
        const float* __restrict__ k, const float* __restrict__ v,
        const float* __restrict__ u, const float* __restrict__ s0,
        float* __restrict__ ys, float* __restrict__ stOut) {
  const int b  = blockIdx.x / Hdim;
  const int h  = blockIdx.x % Hdim;
  const int iz = threadIdx.x >> 6;      // 0..3
  const int j  = threadIdx.x & 63;

  float S[16], uloc[16];
  #pragma unroll
  for (int ii = 0; ii < 16; ++ii) {
    const int i = iz * 16 + ii;
    S[ii]    = s0[((size_t)h * HDs + i) * HDs + j];
    uloc[ii] = u[h * HDs + i];
  }

  __shared__ float vec[4 * 64];     // r | w | k | v
  __shared__ float yred[4][64];

  for (int t = 0; t < Tdim; ++t) {
    const size_t base = ((size_t)(b * Tdim + t)) * Ddim + h * HDs;
    float val;
    if      (iz == 0) val = r[base + j];
    else if (iz == 1) val = w[base + j];
    else if (iz == 2) val = k[base + j];
    else              val = v[base + j];
    vec[iz * 64 + j] = val;
    __syncthreads();

    const float vj = vec[3 * 64 + j];
    float y = 0.f;
    #pragma unroll
    for (int ii = 0; ii < 16; ++ii) {
      const int i = iz * 16 + ii;
      const float kv = vec[2 * 64 + i] * vj;
      y += vec[i] * (S[ii] + uloc[ii] * kv);
      S[ii] = vec[64 + i] * S[ii] + kv;
    }
    yred[iz][j] = y;
    __syncthreads();
    if (iz == 0) ys[base + j] = yred[0][j] + yred[1][j] + yred[2][j] + yred[3][j];
  }

  #pragma unroll
  for (int ii = 0; ii < 16; ++ii) {
    const int i = iz * 16 + ii;
    stOut[(((size_t)b * Hdim + h) * HDs + i) * HDs + j] = S[ii];
  }
}

// ---------------------------------------------------------------- groupnorm(ys) * silu(g) -> bf16
__global__ void __launch_bounds__(256)
tm_gn_gate(const float* __restrict__ ys, const float* __restrict__ g,
           const float* __restrict__ lnw, const float* __restrict__ lnb,
           __bf16* __restrict__ gated) {
  const int p = blockIdx.x * 8 + (threadIdx.x >> 5);   // (token, head) pair
  if (p >= Ntok * Hdim) return;
  const int n = p / Hdim, h = p % Hdim;
  const int lane = threadIdx.x & 31;
  const size_t base = (size_t)n * Ddim + h * HDs;

  const float a = ys[base + lane];
  const float c = ys[base + lane + 32];
  float s = a + c;
  #pragma unroll
  for (int off = 16; off > 0; off >>= 1) s += __shfl_xor(s, off, 32);
  const float mean = s * (1.f / 64.f);
  const float da = a - mean, dc = c - mean;
  float vs = da * da + dc * dc;
  #pragma unroll
  for (int off = 16; off > 0; off >>= 1) vs += __shfl_xor(vs, off, 32);
  const float inv = rsqrtf(vs * (1.f / 64.f) + 1e-5f);

  const float l0 = da * inv * lnw[lane]      + lnb[lane];
  const float l1 = dc * inv * lnw[lane + 32] + lnb[lane + 32];
  const float g0 = g[base + lane];
  const float g1 = g[base + lane + 32];
  gated[base + lane]      = f2bf(g0 / (1.f + expf(-g0)) * l0);
  gated[base + lane + 32] = f2bf(g1 / (1.f + expf(-g1)) * l1);
}

// ---------------------------------------------------------------- host
extern "C" void kernel_launch(void* const* d_in, const int* in_sizes, int n_in,
                              void* d_out, int out_size, void* d_ws, size_t ws_size,
                              hipStream_t stream) {
  (void)in_sizes; (void)n_in; (void)out_size; (void)ws_size;
  const float* x   = (const float*)d_in[0];
  const float* xw  = (const float*)d_in[1];
  const float* Wbig[5] = { (const float*)d_in[2], (const float*)d_in[3],
                           (const float*)d_in[4], (const float*)d_in[5],
                           (const float*)d_in[23] };               // rW kW vW gW oW
  const float *la[5], *lb[5], *llv[5];
  for (int p = 0; p < 5; ++p) {                                    // r k v g d
    la[p]  = (const float*)d_in[6 + 3 * p];
    lb[p]  = (const float*)d_in[7 + 3 * p];
    llv[p] = (const float*)d_in[8 + 3 * p];
  }
  const float* lnw = (const float*)d_in[21];
  const float* lnb = (const float*)d_in[22];
  const float* s0  = (const float*)d_in[24];
  const float* u   = (const float*)d_in[25];
  float* out = (float*)d_out;

  // workspace carve-out
  char* ws = (char*)d_ws;
  size_t cur = 0;
  auto carve = [&](size_t bytes) { void* p = ws + cur; cur += (bytes + 255) & ~(size_t)255; return p; };
  __bf16* xb    = (__bf16*)carve(ND * 2);
  __bf16* dxb   = (__bf16*)carve(ND * 2);
  __bf16* lerpb = (__bf16*)carve(ND * 2);
  __bf16* din   = (__bf16*)carve(ND * 2);
  __bf16* mix[4]; for (int i = 0; i < 4; ++i) mix[i] = (__bf16*)carve(ND * 2);
  __bf16* hbuf  = (__bf16*)carve(NL * 2);
  __bf16* Wb[5]; for (int i = 0; i < 5; ++i) Wb[i] = (__bf16*)carve(DDsz * 2);
  __bf16 *lab[5], *lbb[5];
  for (int i = 0; i < 5; ++i) { lab[i] = (__bf16*)carve(LDsz * 2); lbb[i] = (__bf16*)carve(LDsz * 2); }
  float* rbuf  = (float*)carve(ND * 4);
  float* kbuf  = (float*)carve(ND * 4);
  float* vbuf  = (float*)carve(ND * 4);
  float* gbuf  = (float*)carve(ND * 4);
  float* wbuf  = (float*)carve(ND * 4);
  float* ysbuf = (float*)carve(ND * 4);
  __bf16* gated = mix[0];   // mix buffers are dead once projections finish

  auto conv = [&](const float* s, __bf16* d, long n) {
    tm_f2bf<<<dim3((unsigned)((n + 255) / 256)), dim3(256), 0, stream>>>(s, d, n);
  };
  auto gemm = [&](const __bf16* A, int lda, const __bf16* W, int ldw,
                  int M, int N, int K, int mode, float* oF, __bf16* oB, const float* llp) {
    const int total = (M / 32) * (N / 64);
    tm_wmma_gemm<<<dim3((unsigned)((total + 7) / 8)), dim3(256), 0, stream>>>(
        A, lda, W, ldw, M, N, K, mode, oF, oB, xb, dxb, llp);
  };

  // stage 0: prep + weight conversion
  tm_prep<<<dim3((unsigned)((ND + 255) / 256)), dim3(256), 0, stream>>>(x, xw, xb, dxb, lerpb);
  for (int i = 0; i < 5; ++i) conv(Wbig[i], Wb[i], DDsz);
  for (int i = 0; i < 5; ++i) { conv(la[i], lab[i], LDsz); conv(lb[i], lbb[i], LDsz); }

  // stage 1: LoRA mixes for r,k,v,g
  for (int p = 0; p < 4; ++p) {
    gemm(lerpb, Ddim, lab[p], Ddim, Ntok, LR,   Ddim, MODE_TANH_BF16, nullptr, hbuf,   nullptr);
    gemm(hbuf,  LR,   lbb[p], LR,   Ntok, Ddim, LR,   MODE_MIX_BF16,  nullptr, mix[p], llv[p]);
  }
  // decay channel d: double LoRA, then w = exp(-exp(.))
  gemm(lerpb, Ddim, lab[4], Ddim, Ntok, LR,   Ddim, MODE_TANH_BF16, nullptr, hbuf, nullptr);
  gemm(hbuf,  LR,   lbb[4], LR,   Ntok, Ddim, LR,   MODE_MIX_BF16,  nullptr, din,  llv[4]);
  gemm(din,   Ddim, lab[4], Ddim, Ntok, LR,   Ddim, MODE_TANH_BF16, nullptr, hbuf, nullptr);
  gemm(hbuf,  LR,   lbb[4], LR,   Ntok, Ddim, LR,   MODE_W_F32,     wbuf,    nullptr, llv[4]);

  // stage 2: big projections
  gemm(mix[0], Ddim, Wb[0], Ddim, Ntok, Ddim, Ddim, MODE_F32, rbuf, nullptr, nullptr);
  gemm(mix[1], Ddim, Wb[1], Ddim, Ntok, Ddim, Ddim, MODE_F32, kbuf, nullptr, nullptr);
  gemm(mix[2], Ddim, Wb[2], Ddim, Ntok, Ddim, Ddim, MODE_F32, vbuf, nullptr, nullptr);
  gemm(mix[3], Ddim, Wb[3], Ddim, Ntok, Ddim, Ddim, MODE_F32, gbuf, nullptr, nullptr);

  // stage 3: recurrent scan (writes final state to output tail)
  tm_scan<<<dim3(Bdim * Hdim), dim3(256), 0, stream>>>(rbuf, wbuf, kbuf, vbuf, u, s0,
                                                       ysbuf, out + ND);
  // stage 4: groupnorm + silu gate, then output projection
  tm_gn_gate<<<dim3(Ntok * Hdim / 8), dim3(256), 0, stream>>>(ysbuf, gbuf, lnw, lnb, gated);
  gemm(gated, Ddim, Wb[4], Ddim, Ntok, Ddim, Ddim, MODE_F32, out, nullptr, nullptr);
}
